// BertSelfAttention_22969485099247
// MI455X (gfx1250) — matrix operand order repaired and checked by reading, not compiled, not run on hardware
//
#include <hip/hip_runtime.h>

typedef __attribute__((ext_vector_type(16))) __bf16 v16bf;
typedef __attribute__((ext_vector_type(2)))  __bf16 v2bf;
typedef __attribute__((ext_vector_type(8)))  float  v8f;
typedef __attribute__((ext_vector_type(4)))  unsigned int u32x4;
typedef __attribute__((ext_vector_type(4)))  float  f32x4;

constexpr int Bb = 4, S = 2048, Hdim = 1024, NH = 16, HD = 64;
constexpr int Mrows = Bb * S;   // 8192

union FragBF { v16bf v; u32x4 q[2]; };

// f32 -> bf16 via the native cast; backend picks the best lowering.
__device__ __forceinline__ unsigned pk(float a, float b) {
  v2bf v;
  v[0] = (__bf16)a;
  v[1] = (__bf16)b;
  return __builtin_bit_cast(unsigned, v);
}
__device__ __forceinline__ unsigned short bf16lo(float f) {
  return (unsigned short)(pk(f, f) & 0xffffu);
}

__device__ __forceinline__ void wait_async0() {
#if defined(__AMDGCN__) && __has_builtin(__builtin_amdgcn_s_wait_asynccnt)
  __builtin_amdgcn_s_wait_asynccnt(0);
#else
  asm volatile("s_wait_asynccnt 0x0" ::: "memory");
#endif
}

// 16B-per-lane async DMA: global -> LDS (ASYNCcnt-tracked, GV mode)
__device__ __forceinline__ void async_copy16(unsigned lds_off, const void* gaddr) {
  asm volatile("global_load_async_to_lds_b128 %0, %1, off"
               :: "v"(lds_off), "v"(gaddr) : "memory");
}

// ---------------------------------------------------------------------------
// Kernel 0: one-pass f32 -> bf16 conversion (memory-bound pre-pass so the
// GEMM mainloop is a pure DMA->LDS->WMMA pipeline with no VALU converts).
// ---------------------------------------------------------------------------
__global__ __launch_bounds__(256)
void cvt_bf16_kernel(const float* __restrict__ in,
                     unsigned short* __restrict__ out, int n)
{
  int i = (blockIdx.x * blockDim.x + threadIdx.x) * 8;
  if (i >= n) return;
  f32x4 a = *(const f32x4*)(in + i);
  f32x4 b = *(const f32x4*)(in + i + 4);
  u32x4 d = { pk(a.x, a.y), pk(a.z, a.w), pk(b.x, b.y), pk(b.z, b.w) };
  *(u32x4*)(out + i) = d;
}

// ---------------------------------------------------------------------------
// Kernel 1: fused QKV projection on bf16 inputs.  out = X @ W^T + bias,
// head-split to [B,NH,S,HD] bf16.  Block tile 64(M) x 128(N), 8 waves each
// own a 32x32 tile (2x2 WMMA accumulators), K-step 32.  A/B tiles staged with
// double-buffered GLOBAL_LOAD_ASYNC_TO_LDS_B128 DMA.
// ---------------------------------------------------------------------------
__global__ __launch_bounds__(256)
void qkv_proj_kernel(const unsigned short* __restrict__ xb,
                     const unsigned short* __restrict__ wqb,
                     const unsigned short* __restrict__ wkb,
                     const unsigned short* __restrict__ wvb,
                     const float* __restrict__ bq,
                     const float* __restrict__ bk,
                     const float* __restrict__ bv,
                     unsigned short* __restrict__ qout,
                     unsigned short* __restrict__ kout,
                     unsigned short* __restrict__ vout)
{
  const int z = blockIdx.z;
  const unsigned short* W = (z == 0) ? wqb : (z == 1) ? wkb : wvb;
  const float* bias       = (z == 0) ? bq  : (z == 1) ? bk  : bv;
  unsigned short* out     = (z == 0) ? qout : (z == 1) ? kout : vout;

  // padded rows: 40 elems = 80 B (multiple of 16 -> b128-friendly), x2 buffers
  __shared__ __attribute__((aligned(16))) unsigned short lds_a[2][64 * 40];
  __shared__ __attribute__((aligned(16))) unsigned short lds_b[2][128 * 40];

  const int tid  = threadIdx.x;
  const int lane = tid & 31, wave = tid >> 5;
  const int lo16 = lane & 15, hi = lane >> 4;
  const int wr = wave & 1, wc = wave >> 1;          // 2 x 4 wave grid
  const int m0 = blockIdx.x * 64;
  const int n0 = blockIdx.y * 128;

  const int arow = tid >> 2, akc = (tid & 3) * 8;   // A loader: 64 rows x 32 k
  const int brow = tid >> 1, bkc = (tid & 1) * 16;  // B loader: 128 rows x 32 k

  const unsigned short* asrc = xb + (size_t)(m0 + arow) * Hdim + akc;
  const unsigned short* bsrc = W  + (size_t)(n0 + brow) * Hdim + bkc;

  unsigned a_off[2], b_off[2];
#pragma unroll
  for (int p = 0; p < 2; ++p) {  // generic ptr low 32 bits == LDS offset
    a_off[p] = (unsigned)(unsigned long long)&lds_a[p][arow * 40 + akc];
    b_off[p] = (unsigned)(unsigned long long)&lds_b[p][brow * 40 + bkc];
  }

  // preload tile 0
  async_copy16(a_off[0], asrc);
  async_copy16(b_off[0], bsrc);
  async_copy16(b_off[0] + 16, bsrc + 8);

  v8f acc[2][2] = {};
  constexpr int NIT = Hdim / 32;

  for (int it = 0; it < NIT; ++it) {
    wait_async0();        // this wave's DMA drained
    __syncthreads();      // all waves' DMA for buf[cur] now visible
    const int cur = it & 1;

    if (it + 1 < NIT) {   // kick next tile's DMA into the other buffer
      const int nxt = cur ^ 1;
      const unsigned short* an = asrc + (it + 1) * 32;
      const unsigned short* bn = bsrc + (it + 1) * 32;
      async_copy16(a_off[nxt], an);
      async_copy16(b_off[nxt], bn);
      async_copy16(b_off[nxt] + 16, bn + 8);
    }

    const unsigned short* la = lds_a[cur];
    const unsigned short* lb = lds_b[cur];

    FragBF af[2], bf[2];
#pragma unroll
    for (int i = 0; i < 2; ++i) {          // A frag: lane = row, K chunks {hi*8, 16+hi*8}
      int row = 32 * wr + i * 16 + lo16;
      int c0 = hi * 8;
      af[i].q[0] = *(const u32x4*)&la[row * 40 + c0];
      af[i].q[1] = *(const u32x4*)&la[row * 40 + c0 + 16];
    }
#pragma unroll
    for (int j = 0; j < 2; ++j) {          // B frag: lane = col, K chunks {hi*8, 16+hi*8}
      int col = 32 * wc + j * 16 + lo16;
      int kb = hi * 8;
      bf[j].q[0] = *(const u32x4*)&lb[col * 40 + kb];
      bf[j].q[1] = *(const u32x4*)&lb[col * 40 + kb + 16];
    }
#pragma unroll
    for (int i = 0; i < 2; ++i)
#pragma unroll
      for (int j = 0; j < 2; ++j)
        acc[i][j] = __builtin_amdgcn_wmma_f32_16x16x32_bf16(
            false, af[i].v, false, bf[j].v, (short)0, acc[i][j], false, false);
  }

  // epilogue: + bias, head-split store [B,NH,S,HD] bf16
#pragma unroll
  for (int i = 0; i < 2; ++i) {
#pragma unroll
    for (int j = 0; j < 2; ++j) {
      int n = n0 + 32 * wc + j * 16 + lo16;
      float bias_n = bias[n];
      int h = n >> 6, d = n & 63;
#pragma unroll
      for (int r = 0; r < 8; ++r) {
        int m = m0 + 32 * wr + i * 16 + r + 8 * hi;
        int bidx = m >> 11, s = m & (S - 1);
        float val = acc[i][j][r] + bias_n;
        out[(((size_t)(bidx * NH + h)) * S + s) * HD + d] = bf16lo(val);
      }
    }
  }
}

// ---------------------------------------------------------------------------
// Kernel 2: fused flash attention.  One (b,h) per block; 8 waves x 16 query
// rows = 128 q rows/block; key tiles of 32.  K staged with async global->LDS
// DMA; scores = QK^T/8 + mask, online softmax, ctx += P@V, WMMA bf16.
// ---------------------------------------------------------------------------
__global__ __launch_bounds__(256)
void attn_kernel(const unsigned short* __restrict__ Q,
                 const unsigned short* __restrict__ K,
                 const unsigned short* __restrict__ V,
                 const float* __restrict__ mask,
                 float* __restrict__ out)
{
  __shared__ __attribute__((aligned(16))) unsigned short k_lds[32 * 72]; // [key][d]
  __shared__ __attribute__((aligned(16))) unsigned short v_lds[64 * 40]; // [d][key] (transposed)
  __shared__ __attribute__((aligned(16))) unsigned short p_lds[8 * 16 * 40];

  const int tid = threadIdx.x, lane = tid & 31, wave = tid >> 5;
  const int lo16 = lane & 15, hi = lane >> 4;
  const int h = blockIdx.y, b = blockIdx.z;
  const int q0 = blockIdx.x * 128 + wave * 16;

  const size_t headBase = ((size_t)(b * NH + h)) * S * HD;

  // Q fragments (16 x 64 split into two 16x32 A-frags), loaded once
  FragBF qf[2];
  {
    const unsigned short* qrow = Q + headBase + (size_t)(q0 + lo16) * HD;
#pragma unroll
    for (int j = 0; j < 2; ++j) {
      qf[j].q[0] = *(const u32x4*)(qrow + j * 32 + hi * 8);
      qf[j].q[1] = *(const u32x4*)(qrow + j * 32 + hi * 8 + 16);
    }
  }

  v8f ctx[4] = {};
  float mrow[8], lrow[8];
#pragma unroll
  for (int r = 0; r < 8; ++r) { mrow[r] = -1e30f; lrow[r] = 0.f; }

  const int krow = tid >> 3, kc = (tid & 7) * 8;   // cooperative loader: 32 rows x 64 d
  unsigned short* pw = &p_lds[wave * 16 * 40];

  const unsigned k_lds_off =
      (unsigned)(unsigned long long)&k_lds[krow * 72 + kc];

  for (int kt = 0; kt < S; kt += 32) {
    { // K tile: async DMA global -> LDS, 16B per lane
      const unsigned short* ks = K + headBase + (size_t)(kt + krow) * HD + kc;
      async_copy16(k_lds_off, ks);
      // V tile staged transposed through VGPRs (layout change -> not async-able)
      const unsigned short* vs = V + headBase + (size_t)(kt + krow) * HD + kc;
      alignas(16) unsigned short tmp[8];
      *(u32x4*)tmp = *(const u32x4*)vs;
#pragma unroll
      for (int i = 0; i < 8; ++i)
        v_lds[(kc + i) * 40 + krow] = tmp[i];
    }
    wait_async0();          // drain this wave's async copies
    __syncthreads();        // then make all waves' LDS writes visible

    // scores: two 16x16 tiles covering 32 keys, K-dim = HD = 64 (2 wmma each)
    v8f sacc[2];
#pragma unroll
    for (int c = 0; c < 2; ++c) {
      int key = c * 16 + lo16;
      FragBF kf0, kf1;
      kf0.q[0] = *(const u32x4*)&k_lds[key * 72 + hi * 8];
      kf0.q[1] = *(const u32x4*)&k_lds[key * 72 + hi * 8 + 16];
      kf1.q[0] = *(const u32x4*)&k_lds[key * 72 + 32 + hi * 8];
      kf1.q[1] = *(const u32x4*)&k_lds[key * 72 + 32 + hi * 8 + 16];
      v8f zacc = {};
      zacc = __builtin_amdgcn_wmma_f32_16x16x32_bf16(
          false, qf[0].v, false, kf0.v, (short)0, zacc, false, false);
      zacc = __builtin_amdgcn_wmma_f32_16x16x32_bf16(
          false, qf[1].v, false, kf1.v, (short)0, zacc, false, false);
      sacc[c] = zacc;
    }

    float mv0 = mask[(size_t)b * S + kt + lo16];
    float mv1 = mask[(size_t)b * S + kt + 16 + lo16];

    // online softmax over 32 keys; C-layout: VGPR r, lane -> row r+8*hi, col lo16
#pragma unroll
    for (int r = 0; r < 8; ++r) {
      float s0 = sacc[0][r] * 0.125f + mv0;
      float s1 = sacc[1][r] * 0.125f + mv1;
      float rm = fmaxf(s0, s1);
#pragma unroll
      for (int off = 8; off >= 1; off >>= 1)
        rm = fmaxf(rm, __shfl_xor(rm, off, 32));       // within 16-lane half
      float mn = fmaxf(mrow[r], rm);
      float sc = __expf(mrow[r] - mn);
      float p0 = __expf(s0 - mn);
      float p1 = __expf(s1 - mn);
      float rs = p0 + p1;
#pragma unroll
      for (int off = 8; off >= 1; off >>= 1)
        rs += __shfl_xor(rs, off, 32);
      lrow[r] = lrow[r] * sc + rs;
      mrow[r] = mn;
#pragma unroll
      for (int g = 0; g < 4; ++g) ctx[g][r] *= sc;
      int prow = r + 8 * hi;
      unsigned pp = pk(p0, p1);
      pw[prow * 40 + lo16]      = (unsigned short)(pp & 0xffffu);
      pw[prow * 40 + 16 + lo16] = (unsigned short)(pp >> 16);
    }

    // reload P in A-frag layout (per-wave LDS region; DS ops in-order per wave)
    FragBF pf;
    pf.q[0] = *(const u32x4*)&pw[lo16 * 40 + hi * 8];
    pf.q[1] = *(const u32x4*)&pw[lo16 * 40 + hi * 8 + 16];

    // ctx += P @ V : 4 d-groups of 16
#pragma unroll
    for (int g = 0; g < 4; ++g) {
      FragBF vf;
      vf.q[0] = *(const u32x4*)&v_lds[(g * 16 + lo16) * 40 + hi * 8];
      vf.q[1] = *(const u32x4*)&v_lds[(g * 16 + lo16) * 40 + hi * 8 + 16];
      ctx[g] = __builtin_amdgcn_wmma_f32_16x16x32_bf16(
          false, pf.v, false, vf.v, (short)0, ctx[g], false, false);
    }
    __syncthreads();
  }

  // finalize: ctx / l, merge heads -> out[b, s, h*HD + d] (f32)
#pragma unroll
  for (int g = 0; g < 4; ++g) {
#pragma unroll
    for (int r = 0; r < 8; ++r) {
      int row = q0 + r + 8 * hi;
      int d = g * 16 + lo16;
      out[((size_t)b * S + row) * Hdim + h * HD + d] = ctx[g][r] / lrow[r];
    }
  }
}

extern "C" void kernel_launch(void* const* d_in, const int* in_sizes, int n_in,
                              void* d_out, int out_size, void* d_ws, size_t ws_size,
                              hipStream_t stream) {
  const float* x    = (const float*)d_in[0];
  const float* mask = (const float*)d_in[1];
  const float* wq   = (const float*)d_in[2];
  const float* bq   = (const float*)d_in[3];
  const float* wk   = (const float*)d_in[4];
  const float* bk   = (const float*)d_in[5];
  const float* wv   = (const float*)d_in[6];
  const float* bv   = (const float*)d_in[7];
  float* out = (float*)d_out;

  const size_t qkvElems = (size_t)Bb * NH * S * HD;   // 8.4M
  const size_t xElems   = (size_t)Mrows * Hdim;       // 8.4M
  const size_t wElems   = (size_t)Hdim * Hdim;        // 1M

  unsigned short* qb  = (unsigned short*)d_ws;
  unsigned short* kb  = qb + qkvElems;
  unsigned short* vb  = kb + qkvElems;
  unsigned short* xbf = vb + qkvElems;
  unsigned short* wqb = xbf + xElems;
  unsigned short* wkb = wqb + wElems;
  unsigned short* wvb = wkb + wElems;

  // pre-pass: f32 -> bf16 for activations and weights
  cvt_bf16_kernel<<<dim3(xElems / 2048), 256, 0, stream>>>(x, xbf, (int)xElems);
  cvt_bf16_kernel<<<dim3(wElems / 2048), 256, 0, stream>>>(wq, wqb, (int)wElems);
  cvt_bf16_kernel<<<dim3(wElems / 2048), 256, 0, stream>>>(wk, wkb, (int)wElems);
  cvt_bf16_kernel<<<dim3(wElems / 2048), 256, 0, stream>>>(wv, wvb, (int)wElems);

  dim3 gridP(Mrows / 64, Hdim / 128, 3);
  qkv_proj_kernel<<<gridP, 256, 0, stream>>>(xbf, wqb, wkb, wvb,
                                             bq, bk, bv, qb, kb, vb);

  dim3 gridA(S / 128, NH, Bb);
  attn_kernel<<<gridA, 256, 0, stream>>>(qb, kb, vb, mask, out);
}
